// APPNP_Net_20547123544189
// MI455X (gfx1250) — matrix operand order repaired and checked by reading, not compiled, not run on hardware
//
#include <hip/hip_runtime.h>

// ---------------------------------------------------------------------------
// APPNP on MI455X (gfx1250, wave32):
//   Hmid = relu(x @ W1 + b1)        -- WMMA bf16 GEMM (f32 accum)
//   h    = Hmid @ W2 + b2           -- WMMA bf16 GEMM (f32 accum)
//   gcn_norm degrees (self loops), K=20 gather/scatter propagation steps,
//   log_softmax rows.
// ---------------------------------------------------------------------------

#define ALPHA_F   0.1f
#define ONEMA_F   0.9f
#define KSTEPS    20
#define CDIM      64

typedef __attribute__((ext_vector_type(16))) __bf16        v16bf;
typedef __attribute__((ext_vector_type(8)))  float         v8f;
typedef __attribute__((ext_vector_type(8)))  unsigned int  v8u;

union BfFrag { v8u u; v16bf b; };

__device__ __forceinline__ unsigned short f2bf(float f) {
    unsigned int x = __float_as_uint(f);
    unsigned int r = (x + 0x7FFFu + ((x >> 16) & 1u)) >> 16;  // round-nearest-even
    return (unsigned short)r;
}

// ---------------------------------------------------------------------------
// Tiled GEMM: Out[M,Ncols] = A[M,K] @ B[K,Ncols] + bias (+ optional ReLU)
// Block tile 128x64, K-step 32, 256 threads = 8 waves, wave w -> 16-row strip.
// A/B converted fp32 -> bf16 on LDS store; V_WMMA_F32_16X16X32_BF16.
// ---------------------------------------------------------------------------
template<bool RELU>
__global__ __launch_bounds__(256) void gemm_wmma_bf16(
    const float* __restrict__ A, const float* __restrict__ B,
    const float* __restrict__ bias, float* __restrict__ Out,
    int M, int K, int Ncols)
{
    __shared__ unsigned short lds_a[128 * 40];  // [row][k], pad to 40 elems
    __shared__ unsigned short lds_b[64 * 40];   // [n][k] (transposed), pad 40

    const int tid  = threadIdx.x;
    const int lane = tid & 31;
    const int wid  = tid >> 5;     // 0..7
    const int hx   = lane >> 4;    // lane half (0/1)
    const int l16  = lane & 15;

    const int m0 = blockIdx.x * 128;
    const int n0 = blockIdx.y * 64;

    v8f acc[4];
    for (int t = 0; t < 4; ++t)
        for (int i = 0; i < 8; ++i) acc[t][i] = 0.0f;

    // A-fill coords: 16 contiguous elems per thread
    const int ar = tid >> 1;            // 0..127
    const int ac = (tid & 1) * 16;      // 0 or 16
    // B-fill coords: 8 contiguous N elems per thread of one k-row
    const int bk = tid >> 3;            // 0..31
    const int bn = (tid & 7) * 8;       // 0..56

    for (int k0 = 0; k0 < K; k0 += 32) {
        // --- stage A tile (128x32) ---
        {
            int gr = m0 + ar; if (gr > M - 1) gr = M - 1;   // clamp tail rows
            const float* arow = A + (size_t)gr * K + k0 + ac;
            #pragma unroll
            for (int j = 0; j < 4; ++j) {
                float4 f = ((const float4*)arow)[j];
                unsigned short* d = &lds_a[ar * 40 + ac + j * 4];
                d[0] = f2bf(f.x); d[1] = f2bf(f.y);
                d[2] = f2bf(f.z); d[3] = f2bf(f.w);
            }
        }
        // --- stage B tile (32x64), transposed into [n][k] ---
        {
            const float* brow = B + (size_t)(k0 + bk) * Ncols + n0 + bn;
            #pragma unroll
            for (int j = 0; j < 2; ++j) {
                float4 f = ((const float4*)brow)[j];
                lds_b[(bn + j * 4 + 0) * 40 + bk] = f2bf(f.x);
                lds_b[(bn + j * 4 + 1) * 40 + bk] = f2bf(f.y);
                lds_b[(bn + j * 4 + 2) * 40 + bk] = f2bf(f.z);
                lds_b[(bn + j * 4 + 3) * 40 + bk] = f2bf(f.w);
            }
        }
        __syncthreads();

        // --- A fragment, ISA 16-bit A 16x32 layout (shared by all 4 B tiles) ---
        BfFrag af;
        #pragma unroll
        for (int v = 0; v < 8; ++v) {
            int kp = ((v < 4) ? (2 * v) : (16 + 2 * (v - 4))) + hx * 8;
            af.u[v] = *(const unsigned int*)&lds_a[(wid * 16 + l16) * 40 + kp];
        }
        // --- 4 x (B fragment + WMMA) ---
        #pragma unroll
        for (int t = 0; t < 4; ++t) {
            BfFrag bf;
            #pragma unroll
            for (int v = 0; v < 8; ++v) {
                int kp = ((v < 4) ? (2 * v) : (16 + 2 * (v - 4))) + hx * 8;   // ISA 16-bit B 32x16 layout
                bf.u[v] = *(const unsigned int*)&lds_b[(t * 16 + l16) * 40 + kp];
            }
            acc[t] = __builtin_amdgcn_wmma_f32_16x16x32_bf16(
                false, af.b, false, bf.b, (short)0, acc[t], false, false);
        }
        __syncthreads();
    }

    // --- epilogue: bias (+ReLU), store per ISA f32 C/D layout ---
    #pragma unroll
    for (int t = 0; t < 4; ++t) {
        int col = n0 + t * 16 + l16;
        float bv = bias[col];
        #pragma unroll
        for (int r = 0; r < 8; ++r) {
            int row = m0 + wid * 16 + hx * 8 + r;
            if (row < M) {
                float v = acc[t][r] + bv;
                if (RELU) v = fmaxf(v, 0.0f);
                Out[(size_t)row * Ncols + col] = v;
            }
        }
    }
}

// ---------------------------------------------------------------------------
// Helpers
// ---------------------------------------------------------------------------
__global__ void fill_kernel(float* __restrict__ p, float val, int n) {
    int i = blockIdx.x * blockDim.x + threadIdx.x;
    if (i < n) p[i] = val;
}

__global__ void deg_count(const long long* __restrict__ ei, float* __restrict__ deg, int E) {
    int stride = gridDim.x * blockDim.x;
    for (int e = blockIdx.x * blockDim.x + threadIdx.x; e < E; e += stride)
        atomicAdd(&deg[(int)ei[(size_t)E + e]], 1.0f);
}

__global__ void dinv_kernel(float* __restrict__ deg, int n) {
    int i = blockIdx.x * blockDim.x + threadIdx.x;
    if (i < n) deg[i] = rsqrtf(deg[i]);   // deg >= 1 (self loop)
}

// One wave32 per edge: 32 lanes cover 64 feats as float2; L2-resident atomics.
__global__ __launch_bounds__(256) void edge_prop(
    const long long* __restrict__ ei, const float* __restrict__ dinv,
    const float* __restrict__ z, float* __restrict__ agg, int E)
{
    const int lane   = threadIdx.x & 31;
    const int wave   = blockIdx.x * (blockDim.x >> 5) + (threadIdx.x >> 5);
    const int nwaves = gridDim.x * (blockDim.x >> 5);
    for (int e = wave; e < E; e += nwaves) {
        int src = (int)ei[e];
        int dst = (int)ei[(size_t)E + e];
        float norm = dinv[src] * dinv[dst];
        float2 v = ((const float2*)(z + (size_t)src * CDIM))[lane];
        float* ap = agg + (size_t)dst * CDIM + lane * 2;
        atomicAdd(ap,     v.x * norm);
        atomicAdd(ap + 1, v.y * norm);
    }
}

// z = 0.9*(agg + z*dinv^2) + 0.1*h ; folds self-loop; re-zeros agg for next step
__global__ void mix_kernel(float* __restrict__ z, float* __restrict__ agg,
                           const float* __restrict__ h, const float* __restrict__ dinv,
                           int total)
{
    int idx = blockIdx.x * blockDim.x + threadIdx.x;
    if (idx >= total) return;
    int node = idx >> 6;
    float d = dinv[node];
    float a = agg[idx];
    agg[idx] = 0.0f;
    z[idx] = ONEMA_F * (a + z[idx] * d * d) + ALPHA_F * h[idx];
}

// One wave per node, 2 elems/lane, shfl_xor reductions (wave32)
__global__ __launch_bounds__(256) void logsoftmax_kernel(
    const float* __restrict__ z, float* __restrict__ out, int n)
{
    int lane = threadIdx.x & 31;
    int node = blockIdx.x * (blockDim.x >> 5) + (threadIdx.x >> 5);
    if (node >= n) return;
    float2 v = ((const float2*)(z + (size_t)node * CDIM))[lane];
    float m = fmaxf(v.x, v.y);
    #pragma unroll
    for (int off = 16; off >= 1; off >>= 1) m = fmaxf(m, __shfl_xor(m, off, 32));
    float s = __expf(v.x - m) + __expf(v.y - m);
    #pragma unroll
    for (int off = 16; off >= 1; off >>= 1) s += __shfl_xor(s, off, 32);
    float ls = __logf(s);
    float2 o; o.x = v.x - m - ls; o.y = v.y - m - ls;
    ((float2*)(out + (size_t)node * CDIM))[lane] = o;
}

// ---------------------------------------------------------------------------
extern "C" void kernel_launch(void* const* d_in, const int* in_sizes, int n_in,
                              void* d_out, int out_size, void* d_ws, size_t ws_size,
                              hipStream_t stream)
{
    const float*     x  = (const float*)d_in[0];
    const long long* ei = (const long long*)d_in[1];   // int64 [2,E] row-major
    const float*     W1 = (const float*)d_in[2];
    const float*     b1 = (const float*)d_in[3];
    const float*     W2 = (const float*)d_in[4];
    const float*     b2 = (const float*)d_in[5];
    float*           out = (float*)d_out;

    const int FIN = 512, HID = 256, C = CDIM;
    const int N = in_sizes[0] / FIN;
    const int E = in_sizes[1] / 2;
    const int total = N * C;

    // workspace carve-out (256B aligned)
    char* ws = (char*)d_ws;
    size_t off = 0;
    auto carve = [&](size_t bytes) {
        void* p = ws + off;
        off += (bytes + 255) & ~(size_t)255;
        return p;
    };
    float* h    = (float*)carve((size_t)N * C * 4);
    float* z    = (float*)carve((size_t)N * C * 4);
    float* agg  = (float*)carve((size_t)N * C * 4);
    float* dinv = (float*)carve((size_t)N * 4);
    float* Hmid = (float*)carve((size_t)N * HID * 4);

    // --- MLP encoder (WMMA bf16) ---
    dim3 g1((N + 127) / 128, HID / 64);
    gemm_wmma_bf16<true ><<<g1, 256, 0, stream>>>(x,    W1, b1, Hmid, N, FIN, HID);
    dim3 g2((N + 127) / 128, C / 64);
    gemm_wmma_bf16<false><<<g2, 256, 0, stream>>>(Hmid, W2, b2, h,    N, HID, C);

    // --- gcn_norm degrees (self loops baked in as deg=1) ---
    fill_kernel<<<(N + 255) / 256, 256, 0, stream>>>(dinv, 1.0f, N);
    deg_count<<<2048, 256, 0, stream>>>(ei, dinv, E);
    dinv_kernel<<<(N + 255) / 256, 256, 0, stream>>>(dinv, N);

    // --- init propagation state ---
    fill_kernel<<<(total + 255) / 256, 256, 0, stream>>>(agg, 0.0f, total);
    hipMemcpyAsync(z, h, (size_t)N * C * 4, hipMemcpyDeviceToDevice, stream);

    // --- K propagation steps ---
    for (int k = 0; k < KSTEPS; ++k) {
        edge_prop<<<4096, 256, 0, stream>>>(ei, dinv, z, agg, E);
        mix_kernel<<<(total + 255) / 256, 256, 0, stream>>>(z, agg, h, dinv, total);
    }

    // --- log_softmax ---
    logsoftmax_kernel<<<(N + 7) / 8, 256, 0, stream>>>(z, out, N);
}
